// Space_attention_60670708023350
// MI455X (gfx1250) — matrix operand order repaired and checked by reading, not compiled, not run on hardware
//
#include <hip/hip_runtime.h>
#include <hip/hip_bf16.h>

#define BATCH 2
#define CH    64
#define HWDIM 96
#define NPOS  (HWDIM * HWDIM)   // 9216
#define QT    (NPOS / 16)       // 576 query tiles per batch

typedef __attribute__((ext_vector_type(16))) _Float16 v16h;
typedef __attribute__((ext_vector_type(8)))  _Float16 v8h;
typedef __attribute__((ext_vector_type(8)))  float    v8f;
typedef int v4i __attribute__((vector_size(16)));

typedef __attribute__((address_space(1))) v4i* as1_v4i_p;
typedef __attribute__((address_space(3))) v4i* as3_v4i_p;

union V16 { v16h v; v8h h[2]; };

__device__ __forceinline__ v8f wmma_f16(v16h a, v16h b, v8f c) {
    // D = A(16x32 f16) * B(32x16 f16) + C(16x16 f32)
    return __builtin_amdgcn_wmma_f32_16x16x32_f16(
        /*neg_a=*/false, a, /*neg_b=*/false, b,
        /*c_mod=*/(short)0, c, /*reuse_a=*/false, /*reuse_b=*/false);
}

// ---- gfx1250 async global->LDS copy (ASYNCcnt-tracked), with fallback ----
#if __has_builtin(__builtin_amdgcn_global_load_async_to_lds_b128)
#define HAVE_ASYNC_LDS 1
#else
#define HAVE_ASYNC_LDS 0
#endif

__device__ __forceinline__ void async_cp16(const _Float16* g, _Float16* l) {
#if HAVE_ASYNC_LDS
    __builtin_amdgcn_global_load_async_to_lds_b128(
        (as1_v4i_p)(unsigned long long)(g),
        (as3_v4i_p)(l), 0, 0);
#else
    *(v8h*)l = *(const v8h*)g;          // register bounce + ds_store_b128
#endif
}

__device__ __forceinline__ void wait_async_zero() {
#if HAVE_ASYNC_LDS
#if __has_builtin(__builtin_amdgcn_s_wait_asynccnt)
    __builtin_amdgcn_s_wait_asynccnt(0);
#else
    asm volatile("s_wait_asynccnt 0" ::: "memory");
#endif
#endif
}

// ---- DPP16 lane permutes (VALU, replaces ds_bpermute shuffles) ----
template <int CTRL>
__device__ __forceinline__ float dppf(float v) {
    union { float f; int i; } u;
    u.f = v;
    u.i = __builtin_amdgcn_update_dpp(u.i, u.i, CTRL, 0xF, 0xF, false);
    return u.f;
}

// butterfly reduction across the 16-lane half-wave (stays inside each
// 16-lane row; matches the WMMA C-fragment row split)
__device__ __forceinline__ float redmax16(float v) {
    v = fmaxf(v, dppf<0xB1>(v));    // quad_perm [1,0,3,2]  (xor 1)
    v = fmaxf(v, dppf<0x4E>(v));    // quad_perm [2,3,0,1]  (xor 2)
    v = fmaxf(v, dppf<0x141>(v));   // row_half_mirror      (xor 7)
    v = fmaxf(v, dppf<0x140>(v));   // row_mirror           (xor 15)
    return v;
}
__device__ __forceinline__ float redsum16(float v) {
    v += dppf<0xB1>(v);
    v += dppf<0x4E>(v);
    v += dppf<0x141>(v);
    v += dppf<0x140>(v);
    return v;
}

// ---------------------------------------------------------------------------
// Kernel 1: 1x1-conv projections.
//   Kq[b][n][o] = Kw . x  (+Kb)   queries, fp16, [B,N,C]
//   Kk[b][n][o] = Qw . x  (+Qb)   keys,    fp16, [B,N,C]
//   Vv[b][o][n] = Vw . y  (+Vb)   values,  fp16, [B,C,N]
// Weight reads use uniform addresses -> scalar loads through constant cache.
// ---------------------------------------------------------------------------
__global__ __launch_bounds__(256) void proj_kernel(
    const float* __restrict__ x,  const float* __restrict__ y,
    const float* __restrict__ Kw, const float* __restrict__ Kb,
    const float* __restrict__ Qw, const float* __restrict__ Qb,
    const float* __restrict__ Vw, const float* __restrict__ Vb,
    _Float16* __restrict__ Kq, _Float16* __restrict__ Kk,
    _Float16* __restrict__ Vv)
{
    const int t = blockIdx.x * 256 + threadIdx.x;   // [0, B*N)
    const int b = t / NPOS;
    const int n = t % NPOS;

    float xr[CH];
    const float* xb = x + (size_t)b * CH * NPOS + n;
    #pragma unroll
    for (int c = 0; c < CH; ++c) xr[c] = xb[(size_t)c * NPOS];  // coalesced per c

    _Float16* kq = Kq + ((size_t)b * NPOS + n) * CH;
    _Float16* kk = Kk + ((size_t)b * NPOS + n) * CH;
    for (int o = 0; o < CH; ++o) {
        float aK = Kb[o];
        float aQ = Qb[o];
        #pragma unroll
        for (int c = 0; c < CH; ++c) {
            aK = fmaf(Kw[o * CH + c], xr[c], aK);
            aQ = fmaf(Qw[o * CH + c], xr[c], aQ);
        }
        kq[o] = (_Float16)aK;
        kk[o] = (_Float16)aQ;
    }

    const float* yb = y + (size_t)b * CH * NPOS + n;
    #pragma unroll
    for (int c = 0; c < CH; ++c) xr[c] = yb[(size_t)c * NPOS];

    _Float16* vv = Vv + (size_t)b * CH * NPOS + n;
    for (int o = 0; o < CH; ++o) {
        float aV = Vb[o];
        #pragma unroll
        for (int c = 0; c < CH; ++c) aV = fmaf(Vw[o * CH + c], xr[c], aV);
        vv[(size_t)o * NPOS] = (_Float16)aV;   // coalesced along n
    }
}

// ---------------------------------------------------------------------------
// Kernel 2: flash attention.
//   * 8 waves / block, one wave owns 16 query rows; all waves of a block
//     share the same 64-key K/V tiles.
//   * Tiles staged into LDS ONCE per block with gfx1250 async global->LDS
//     copies (ASYNCcnt), double-buffered: next tile streams while current
//     one feeds the WMMAs.  8x L2-traffic reduction vs per-wave streaming.
//   * Per 64 keys: 8x v_wmma for S, DPP16 online softmax, P (16x64 fp16)
//     bounced through per-wave LDS (C-layout -> A-layout), 8x v_wmma for PV.
// ---------------------------------------------------------------------------
#define TPITCH 72   // halves per staged row: 144B, 16B-aligned, bank-spread

__global__ __launch_bounds__(256) void attn_kernel(
    const _Float16* __restrict__ Kq,   // queries [B,N,C]
    const _Float16* __restrict__ Kk,   // keys    [B,N,C]
    const _Float16* __restrict__ Vv,   // values  [B,C,N]
    float* __restrict__ Of)            // output  [B,N,C] f32
{
    __shared__ _Float16 Ktile[2][64 * TPITCH];   // keys:   [key][ch]
    __shared__ _Float16 Vtile[2][64 * TPITCH];   // values: [ch][key]
    __shared__ _Float16 Plds[8 * 16 * TPITCH];   // per-wave P tiles

    const int tid  = threadIdx.x;
    const int lane = tid & 31;
    const int wv   = tid >> 5;
    const int b    = (blockIdx.x * 8) / QT;          // uniform per block
    const int q0   = ((blockIdx.x * 8 + wv) % QT) * 16;
    const int hl   = lane >> 4;                      // half-wave: 0 or 1
    const int l16  = lane & 15;

    const _Float16* Qp = Kq + (size_t)b * NPOS * CH;
    const _Float16* Kp = Kk + (size_t)b * NPOS * CH;
    const _Float16* Vp = Vv + (size_t)b * CH * NPOS;

    // cooperative stage of one 64-key K tile + V tile (2x 8KB) into LDS:
    // 512 chunks of 16B per tile, 2 chunks/thread/tile
    auto stage = [&](int buf, int j0) {
        #pragma unroll
        for (int i = 0; i < 2; ++i) {
            const int c   = tid * 2 + i;      // keep a thread's chunks adjacent
            const int row = c >> 3;           // 0..63
            const int off = (c & 7) * 8;      // halves within row
            async_cp16(Kp + (size_t)(j0 + row) * CH + off,
                       &Ktile[buf][row * TPITCH + off]);
            async_cp16(Vp + (size_t)row * NPOS + j0 + off,
                       &Vtile[buf][row * TPITCH + off]);
        }
    };

    // A fragment of the 16 query rows (16x32 f16 layout: lane=row,
    // lanes 0-15 hold K 0-7 & 16-23, lanes 16-31 hold K 8-15 & 24-31).
    const _Float16* qrow = Qp + (size_t)(q0 + l16) * CH + hl * 8;
    V16 aQ0, aQ1;
    aQ0.h[0] = *(const v8h*)(qrow +  0);   // ch [hl*8,    +8)
    aQ0.h[1] = *(const v8h*)(qrow + 16);   // ch [hl*8+16, +8)
    aQ1.h[0] = *(const v8h*)(qrow + 32);   // ch 32..63 fragment
    aQ1.h[1] = *(const v8h*)(qrow + 48);

    v8f acc[4] = {};                        // 16 rows x 64 channels
    float mrun[8], lrun[8];
    #pragma unroll
    for (int r = 0; r < 8; ++r) { mrun[r] = -3.0e38f; lrun[r] = 0.0f; }

    _Float16* Pw = Plds + wv * (16 * TPITCH);

    stage(0, 0);
    wait_async_zero();
    __syncthreads();

    int p = 0;
    #pragma unroll 1
    for (int j0 = 0; j0 < NPOS; j0 += 64) {
        if (j0 + 64 < NPOS) stage(p ^ 1, j0 + 64);   // uniform branch

        const _Float16* Kt = Ktile[p];
        const _Float16* Vt = Vtile[p];

        // ---- S = Q . K^T for keys [j0, j0+64): four 16x16 f32 tiles ----
        v8f s[4];
        #pragma unroll
        for (int sb = 0; sb < 4; ++sb) {
            const _Float16* krow = Kt + (sb * 16 + l16) * TPITCH + hl * 8;
            V16 bK;
            v8f acc_s = {};
            bK.h[0] = *(const v8h*)(krow + 0);    // ch [hl*8, +8) & [hl*8+16, +8)
            bK.h[1] = *(const v8h*)(krow + 16);
            acc_s = wmma_f16(aQ0.v, bK.v, acc_s);
            bK.h[0] = *(const v8h*)(krow + 32);   // ch 32..63
            bK.h[1] = *(const v8h*)(krow + 48);
            s[sb] = wmma_f16(aQ1.v, bK.v, acc_s);
        }

        // ---- joint online softmax over these 64 logits per row ----
        // C-layout: VGPR r holds row (r + 8*hl), col = l16 within half-wave.
        #pragma unroll
        for (int r = 0; r < 8; ++r) {
            float mx = fmaxf(fmaxf(s[0][r], s[1][r]),
                             fmaxf(s[2][r], s[3][r]));
            mx = redmax16(mx);
            float mo = mrun[r];
            float mn = fmaxf(mo, mx);
            float sc = __expf(mo - mn);
            float p0 = __expf(s[0][r] - mn);
            float p1 = __expf(s[1][r] - mn);
            float p2 = __expf(s[2][r] - mn);
            float p3 = __expf(s[3][r] - mn);
            float ps = redsum16((p0 + p1) + (p2 + p3));
            lrun[r] = lrun[r] * sc + ps;
            mrun[r] = mn;
            acc[0][r] *= sc; acc[1][r] *= sc; acc[2][r] *= sc; acc[3][r] *= sc;

            const int row = r + hl * 8;
            _Float16* pr = Pw + row * TPITCH + l16;
            pr[0]  = (_Float16)p0;
            pr[16] = (_Float16)p1;
            pr[32] = (_Float16)p2;
            pr[48] = (_Float16)p3;
        }

        // ---- reload P as two 16x32 f16 A fragments ----
        const _Float16* prow = Pw + l16 * TPITCH + hl * 8;
        V16 aP0, aP1;
        aP0.h[0] = *(const v8h*)(prow + 0);
        aP0.h[1] = *(const v8h*)(prow + 16);
        aP1.h[0] = *(const v8h*)(prow + 32);
        aP1.h[1] = *(const v8h*)(prow + 48);

        // ---- O += P . V  (4 channel tiles x 2 K-steps of 32 keys) ----
        #pragma unroll
        for (int tile = 0; tile < 4; ++tile) {
            // B 32x16: lane = channel (tile*16 + l16), keys hl*8+[0,8) & hl*8+16+[0,8)
            const _Float16* vcol = Vt + (tile * 16 + l16) * TPITCH + hl * 8;
            V16 bV;
            bV.h[0] = *(const v8h*)(vcol + 0);     // keys j0    .. +32
            bV.h[1] = *(const v8h*)(vcol + 16);
            acc[tile] = wmma_f16(aP0.v, bV.v, acc[tile]);
            bV.h[0] = *(const v8h*)(vcol + 32);    // keys j0+32 .. +64
            bV.h[1] = *(const v8h*)(vcol + 48);
            acc[tile] = wmma_f16(aP1.v, bV.v, acc[tile]);
        }

        wait_async_zero();   // next tile landed
        __syncthreads();     // and everyone is done reading tile p
        p ^= 1;
    }

    // ---- normalize and write O[b][q0+row][*] (f32) ----
    float* Ob = Of + ((size_t)b * NPOS + q0) * CH;
    #pragma unroll
    for (int r = 0; r < 8; ++r) {
        float inv = 1.0f / lrun[r];
        int row = r + hl * 8;
        #pragma unroll
        for (int tile = 0; tile < 4; ++tile)
            Ob[(size_t)row * CH + tile * 16 + l16] = acc[tile][r] * inv;
    }
}

// ---------------------------------------------------------------------------
// Kernel 3: out = y + conv1x1(O, Lw, Lb).  Per-(b,n) thread, O row is
// contiguous; weights via uniform scalar loads; coalesced stores along n.
// ---------------------------------------------------------------------------
__global__ __launch_bounds__(256) void out_kernel(
    const float* __restrict__ Of, const float* __restrict__ y,
    const float* __restrict__ Lw, const float* __restrict__ Lb,
    float* __restrict__ out)
{
    const int t = blockIdx.x * 256 + threadIdx.x;
    const int b = t / NPOS;
    const int n = t % NPOS;

    float orow[CH];
    const float* op = Of + ((size_t)b * NPOS + n) * CH;
    #pragma unroll
    for (int c = 0; c < CH; ++c) orow[c] = op[c];

    const float* yb = y   + (size_t)b * CH * NPOS + n;
    float*       ob = out + (size_t)b * CH * NPOS + n;
    for (int o = 0; o < CH; ++o) {
        float a = Lb[o];
        #pragma unroll
        for (int c = 0; c < CH; ++c) a = fmaf(Lw[o * CH + c], orow[c], a);
        ob[(size_t)o * NPOS] = yb[(size_t)o * NPOS] + a;
    }
}

// ---------------------------------------------------------------------------
extern "C" void kernel_launch(void* const* d_in, const int* in_sizes, int n_in,
                              void* d_out, int out_size, void* d_ws, size_t ws_size,
                              hipStream_t stream) {
    const float* x  = (const float*)d_in[0];
    const float* y  = (const float*)d_in[1];
    const float* Kw = (const float*)d_in[2];
    const float* Kb = (const float*)d_in[3];
    const float* Qw = (const float*)d_in[4];
    const float* Qb = (const float*)d_in[5];
    const float* Vw = (const float*)d_in[6];
    const float* Vb = (const float*)d_in[7];
    const float* Lw = (const float*)d_in[8];
    const float* Lb = (const float*)d_in[9];

    const size_t nElem = (size_t)BATCH * NPOS * CH;    // 1,179,648
    _Float16* Kq = (_Float16*)d_ws;                    // queries [B,N,C] fp16
    _Float16* Kk = Kq + nElem;                         // keys    [B,N,C] fp16
    _Float16* Vv = Kk + nElem;                         // values  [B,C,N] fp16
    float*    Of = (float*)(Vv + nElem);               // attn out [B,N,C] f32
    // total ws usage: 3*2*nElem + 4*nElem = 11,796,480 bytes

    proj_kernel<<<(BATCH * NPOS) / 256, 256, 0, stream>>>(
        x, y, Kw, Kb, Qw, Qb, Vw, Vb, Kq, Kk, Vv);

    attn_kernel<<<(BATCH * NPOS / 16) / 8, 256, 0, stream>>>(Kq, Kk, Vv, Of);

    out_kernel<<<(BATCH * NPOS) / 256, 256, 0, stream>>>(Of, y, Lw, Lb,
                                                         (float*)d_out);
}
